// SpikingSiameseNetwork_66168266162719
// MI455X (gfx1250) — compile-verified, matrix operand order checked by reading, not run on hardware
//
#include <hip/hip_runtime.h>
#include <math.h>

// ---------------------------------------------------------------------------
// Spiking Siamese net, MI455X (gfx1250, wave32, WMMA 16x16x32 f16)
// B=4096 T=25 AD=768 VD=512 H=256 E=128
// 128 blocks x 256 threads; each block owns 32 batch rows (2 x 16-row M-tiles)
// for all 25 steps -> every weight B-fragment loaded from L2 feeds 2 WMMAs.
// Membrane potentials live in v8f WMMA accumulators (fp32) for the whole run.
// K-loops at unroll 2: double-buffered fragment registers remove the WAR
// serialization between iteration i's WMMAs and iteration i+1's loads.
// ---------------------------------------------------------------------------

#define B_TOT   4096
#define T_STEPS 25
#define ROWS    32           // batch rows per block (2 M-subtiles)
#define AD      768
#define VD      512
#define H_DIM   256
#define E_DIM   128
#define LDA     (AD + 8)     // halfs; padded so rows hit distinct LDS bank groups
#define LDS_S   (H_DIM + 8)  // spike stage stride (halfs)
#define BETA    0.9f
#define THRESH  1.0f

typedef __attribute__((ext_vector_type(16))) _Float16 v16h;
typedef __attribute__((ext_vector_type(8)))  _Float16 v8h;
typedef __attribute__((ext_vector_type(4)))  _Float16 v4h;
typedef __attribute__((ext_vector_type(8)))  float    v8f;

// ---------------------------------------------------------------------------
// Pre-pass: convert fp32 weight [N][K] (out x in) into f16 WMMA B-fragments.
// B[k][n] = W[n][k].  Fragment for (tile = n>>4, kstep = k>>5) is 512 halfs:
// index = lane*16 + j, where for kk = k&31:
//   lane = (n&15) | (((kk>>3)&1) << 4)
//   j    = (kk&7) | (((kk>>4)&1) << 3)
// ---------------------------------------------------------------------------
__global__ void snn_convert_weights(const float* __restrict__ W,
                                    _Float16* __restrict__ frag,
                                    int N, int K) {
  int idx = blockIdx.x * blockDim.x + threadIdx.x;
  if (idx >= N * K) return;
  int n = idx / K, k = idx % K;
  int tile = n >> 4, kstep = k >> 5;
  int kk = k & 31;
  int lane = (n & 15) | (((kk >> 3) & 1) << 4);
  int j    = (kk & 7) | (((kk >> 4) & 1) << 3);
  int ksteps = K >> 5;
  frag[((size_t)(tile * ksteps + kstep) << 9) + (lane << 4) + j] = (_Float16)W[idx];
}

__device__ __forceinline__ v8f splat8(float x) {
  v8f v = {x, x, x, x, x, x, x, x};
  return v;
}

// A-fragment (16-bit 16x32 layout): row m = lane&15, half-select = lane>>4.
__device__ __forceinline__ v16h load_afrag(const _Float16* base, int stride,
                                           int lane, int kb) {
  const _Float16* p = base + (lane & 15) * stride + kb;
  v8h lo = *(const v8h*)p;
  v8h hi = *(const v8h*)(p + 16);
  return __builtin_shufflevector(lo, hi, 0, 1, 2, 3, 4, 5, 6, 7,
                                 8, 9, 10, 11, 12, 13, 14, 15);
}

// LIF update (snntorch Leaky, reset-by-subtract) + f16 spike store to LDS.
// mrow0: LDS row base for this M-subtile.
__device__ __forceinline__ void lif_spike_lds(v8f& mem, const v8f& x,
                                              _Float16* sbase, int stride,
                                              int mrow0, int khalf, int col) {
#pragma unroll
  for (int r = 0; r < 8; ++r) {
    float mm = mem[r];
    float reset = (mm > THRESH) ? 1.0f : 0.0f;
    mm = BETA * mm + x[r] - reset * THRESH;
    mem[r] = mm;
    sbase[(mrow0 + r + khalf) * stride + col] =
        (_Float16)(((mm - THRESH) > 0.0f) ? 1.0f : 0.0f);
  }
}

// LIF update for layer-2 tiles; writes fp32 spike to ldsOut only at last step.
__device__ __forceinline__ void lif_out(v8f& mem, const v8f& x, bool last,
                                        float (*obase)[2 * E_DIM],
                                        int mrow0, int khalf, int col) {
#pragma unroll
  for (int r = 0; r < 8; ++r) {
    float mm = mem[r];
    float reset = (mm > THRESH) ? 1.0f : 0.0f;
    mm = BETA * mm + x[r] - reset * THRESH;
    mem[r] = mm;
    if (last)
      obase[mrow0 + r + khalf][col] = ((mm - THRESH) > 0.0f) ? 1.0f : 0.0f;
  }
}

__global__ __launch_bounds__(256, 1) void snn_kernel(
    const float* __restrict__ a_spk, const float* __restrict__ v_spk,
    const _Float16* __restrict__ Wa1f, const float* __restrict__ ba1,
    const _Float16* __restrict__ Wa2f, const float* __restrict__ ba2,
    const _Float16* __restrict__ Wv1f, const float* __restrict__ bv1,
    const _Float16* __restrict__ Wv2f, const float* __restrict__ bv2,
    float* __restrict__ out) {
  __shared__ _Float16 ldsA[ROWS * LDA];          // input stage (a or v), f16
  __shared__ _Float16 ldsS[ROWS * LDS_S];        // layer-1 spike stage, f16
  __shared__ _Float16 ldsW2[2 * 8 * 8 * 512];    // layer-2 B-fragments (a|v)
  __shared__ float    ldsOut[ROWS][2 * E_DIM];   // final spikes
  __shared__ float    ldsRed[ROWS][8];
  __shared__ float    ldsNorm[ROWS];

  const int tid   = threadIdx.x;
  const int wave  = tid >> 5;
  const int lane  = tid & 31;
  const int row0  = blockIdx.x * ROWS;
  const int n16   = lane & 15;
  const int khalf = (lane >> 4) << 3;

  // --- cache layer-2 weight fragments in LDS (reused all 25 steps) ---
  for (int i = tid; i < (8 * 8 * 512) / 8; i += 256)
    ((v8h*)ldsW2)[i] = ((const v8h*)Wa2f)[i];
  for (int i = tid; i < (8 * 8 * 512) / 8; i += 256)
    ((v8h*)(ldsW2 + 8 * 8 * 512))[i] = ((const v8h*)Wv2f)[i];

  // --- per-lane biases (C-layout column = lane&15); shared by both M-subtiles
  const float bia1_0 = ba1[wave * 32 + n16];
  const float bia1_1 = ba1[wave * 32 + 16 + n16];
  const float biv1_0 = bv1[wave * 32 + n16];
  const float biv1_1 = bv1[wave * 32 + 16 + n16];
  const float bia2   = ba2[wave * 16 + n16];
  const float biv2   = bv2[wave * 16 + n16];

  // --- persistent membrane potentials (fp32, VGPR-resident) ---
  // layer1: [msub][ntile]; layer2: [msub]
  v8f ma1_00 = splat8(0.f), ma1_01 = splat8(0.f);
  v8f ma1_10 = splat8(0.f), ma1_11 = splat8(0.f);
  v8f mv1_00 = splat8(0.f), mv1_01 = splat8(0.f);
  v8f mv1_10 = splat8(0.f), mv1_11 = splat8(0.f);
  v8f ma2_0 = splat8(0.f), ma2_1 = splat8(0.f);
  v8f mv2_0 = splat8(0.f), mv2_1 = splat8(0.f);

  // per-lane fragment base pointers (layer-1 tiles 2w, 2w+1)
  const _Float16* bpA  = Wa1f + (((size_t)(wave * 2) * (AD / 32)) << 9) + (lane << 4);
  const _Float16* bpV  = Wv1f + (((size_t)(wave * 2) * (VD / 32)) << 9) + (lane << 4);
  const _Float16* bpA2 = ldsW2 + ((wave * 8) << 9) + (lane << 4);
  const _Float16* bpV2 = ldsW2 + 8 * 8 * 512 + ((wave * 8) << 9) + (lane << 4);

  __syncthreads();

  for (int t = 0; t < T_STEPS; ++t) {
    const bool last = (t == T_STEPS - 1);

    // ============== audio branch ==============
    // stage a_t tile [32 x 768] fp32 -> f16 LDS
    for (int i = tid; i < ROWS * (AD / 4); i += 256) {
      int r = i / (AD / 4), c = i % (AD / 4);
      float4 f = ((const float4*)(a_spk + ((size_t)(row0 + r) * T_STEPS + t) * AD))[c];
      v4h h = {(_Float16)f.x, (_Float16)f.y, (_Float16)f.z, (_Float16)f.w};
      *(v4h*)&ldsA[r * LDA + c * 4] = h;
    }
    __syncthreads();

    // layer a1: 2 N-tiles x 2 M-subtiles per wave; B-frags reused across M
    {
      v8f a00 = splat8(bia1_0), a01 = splat8(bia1_1);
      v8f a10 = splat8(bia1_0), a11 = splat8(bia1_1);
#pragma unroll 2
      for (int ks = 0; ks < AD / 32; ++ks) {
        int kb = khalf + ks * 32;
        v16h af0 = load_afrag(ldsA, LDA, lane, kb);
        v16h af1 = load_afrag(ldsA + 16 * LDA, LDA, lane, kb);
        v16h bf0 = *(const v16h*)(bpA + ((size_t)ks << 9));
        v16h bf1 = *(const v16h*)(bpA + ((size_t)(AD / 32 + ks) << 9));
        a00 = __builtin_amdgcn_wmma_f32_16x16x32_f16(false, af0, false, bf0,
                                                     (short)0, a00, false, false);
        a01 = __builtin_amdgcn_wmma_f32_16x16x32_f16(false, af0, false, bf1,
                                                     (short)0, a01, false, false);
        a10 = __builtin_amdgcn_wmma_f32_16x16x32_f16(false, af1, false, bf0,
                                                     (short)0, a10, false, false);
        a11 = __builtin_amdgcn_wmma_f32_16x16x32_f16(false, af1, false, bf1,
                                                     (short)0, a11, false, false);
      }
      lif_spike_lds(ma1_00, a00, ldsS, LDS_S, 0,  khalf, (wave * 2) * 16 + n16);
      lif_spike_lds(ma1_01, a01, ldsS, LDS_S, 0,  khalf, (wave * 2 + 1) * 16 + n16);
      lif_spike_lds(ma1_10, a10, ldsS, LDS_S, 16, khalf, (wave * 2) * 16 + n16);
      lif_spike_lds(ma1_11, a11, ldsS, LDS_S, 16, khalf, (wave * 2 + 1) * 16 + n16);
    }
    __syncthreads();

    // layer a2: 1 N-tile x 2 M-subtiles per wave, K=256, B from LDS cache
    {
      v8f a0 = splat8(bia2), a1 = splat8(bia2);
#pragma unroll 2
      for (int ks = 0; ks < H_DIM / 32; ++ks) {
        int kb = khalf + ks * 32;
        v16h af0 = load_afrag(ldsS, LDS_S, lane, kb);
        v16h af1 = load_afrag(ldsS + 16 * LDS_S, LDS_S, lane, kb);
        v16h bf = *(const v16h*)(bpA2 + ((size_t)ks << 9));
        a0 = __builtin_amdgcn_wmma_f32_16x16x32_f16(false, af0, false, bf,
                                                    (short)0, a0, false, false);
        a1 = __builtin_amdgcn_wmma_f32_16x16x32_f16(false, af1, false, bf,
                                                    (short)0, a1, false, false);
      }
      lif_out(ma2_0, a0, last, ldsOut, 0,  khalf, wave * 16 + n16);
      lif_out(ma2_1, a1, last, ldsOut, 16, khalf, wave * 16 + n16);
    }

    // ============== visual branch ==============
    for (int i = tid; i < ROWS * (VD / 4); i += 256) {
      int r = i / (VD / 4), c = i % (VD / 4);
      float4 f = ((const float4*)(v_spk + ((size_t)(row0 + r) * T_STEPS + t) * VD))[c];
      v4h h = {(_Float16)f.x, (_Float16)f.y, (_Float16)f.z, (_Float16)f.w};
      *(v4h*)&ldsA[r * LDA + c * 4] = h;
    }
    __syncthreads();

    {
      v8f a00 = splat8(biv1_0), a01 = splat8(biv1_1);
      v8f a10 = splat8(biv1_0), a11 = splat8(biv1_1);
#pragma unroll 2
      for (int ks = 0; ks < VD / 32; ++ks) {
        int kb = khalf + ks * 32;
        v16h af0 = load_afrag(ldsA, LDA, lane, kb);
        v16h af1 = load_afrag(ldsA + 16 * LDA, LDA, lane, kb);
        v16h bf0 = *(const v16h*)(bpV + ((size_t)ks << 9));
        v16h bf1 = *(const v16h*)(bpV + ((size_t)(VD / 32 + ks) << 9));
        a00 = __builtin_amdgcn_wmma_f32_16x16x32_f16(false, af0, false, bf0,
                                                     (short)0, a00, false, false);
        a01 = __builtin_amdgcn_wmma_f32_16x16x32_f16(false, af0, false, bf1,
                                                     (short)0, a01, false, false);
        a10 = __builtin_amdgcn_wmma_f32_16x16x32_f16(false, af1, false, bf0,
                                                     (short)0, a10, false, false);
        a11 = __builtin_amdgcn_wmma_f32_16x16x32_f16(false, af1, false, bf1,
                                                     (short)0, a11, false, false);
      }
      lif_spike_lds(mv1_00, a00, ldsS, LDS_S, 0,  khalf, (wave * 2) * 16 + n16);
      lif_spike_lds(mv1_01, a01, ldsS, LDS_S, 0,  khalf, (wave * 2 + 1) * 16 + n16);
      lif_spike_lds(mv1_10, a10, ldsS, LDS_S, 16, khalf, (wave * 2) * 16 + n16);
      lif_spike_lds(mv1_11, a11, ldsS, LDS_S, 16, khalf, (wave * 2 + 1) * 16 + n16);
    }
    __syncthreads();

    {
      v8f a0 = splat8(biv2), a1 = splat8(biv2);
#pragma unroll 2
      for (int ks = 0; ks < H_DIM / 32; ++ks) {
        int kb = khalf + ks * 32;
        v16h af0 = load_afrag(ldsS, LDS_S, lane, kb);
        v16h af1 = load_afrag(ldsS + 16 * LDS_S, LDS_S, lane, kb);
        v16h bf = *(const v16h*)(bpV2 + ((size_t)ks << 9));
        a0 = __builtin_amdgcn_wmma_f32_16x16x32_f16(false, af0, false, bf,
                                                    (short)0, a0, false, false);
        a1 = __builtin_amdgcn_wmma_f32_16x16x32_f16(false, af1, false, bf,
                                                    (short)0, a1, false, false);
      }
      lif_out(mv2_0, a0, last, ldsOut, 0,  khalf, E_DIM + wave * 16 + n16);
      lif_out(mv2_1, a1, last, ldsOut, 16, khalf, E_DIM + wave * 16 + n16);
    }
  } // timestep loop

  // ---- row-wise L2 normalization of [32 x 256] and write-out ----
  __syncthreads();
  {
    int r = tid >> 3, seg = tid & 7;  // 32 rows x 8 segments of 32 cols
    float s = 0.f;
#pragma unroll
    for (int c = 0; c < 32; ++c) {
      float x = ldsOut[r][seg * 32 + c];
      s += x * x;
    }
    ldsRed[r][seg] = s;
  }
  __syncthreads();
  if (tid < ROWS) {
    float s = 0.f;
#pragma unroll
    for (int i = 0; i < 8; ++i) s += ldsRed[tid][i];
    float n = sqrtf(s);
    ldsNorm[tid] = 1.0f / fmaxf(n, 1e-12f);
  }
  __syncthreads();
  for (int i = tid; i < ROWS * 2 * E_DIM; i += 256) {
    int r = i >> 8, c = i & 255;
    out[(size_t)(row0 + r) * (2 * E_DIM) + c] = ldsOut[r][c] * ldsNorm[r];
  }
}

// ---------------------------------------------------------------------------
extern "C" void kernel_launch(void* const* d_in, const int* in_sizes, int n_in,
                              void* d_out, int out_size, void* d_ws, size_t ws_size,
                              hipStream_t stream) {
  (void)in_sizes; (void)n_in; (void)out_size; (void)ws_size;
  const float* a_spk = (const float*)d_in[0];
  const float* v_spk = (const float*)d_in[1];
  const float* Wa1 = (const float*)d_in[2];
  const float* ba1 = (const float*)d_in[3];
  const float* Wa2 = (const float*)d_in[4];
  const float* ba2 = (const float*)d_in[5];
  const float* Wv1 = (const float*)d_in[6];
  const float* bv1 = (const float*)d_in[7];
  const float* Wv2 = (const float*)d_in[8];
  const float* bv2 = (const float*)d_in[9];

  // workspace layout (halfs): Wa1 fragments | Wa2 | Wv1 | Wv2  (786 KB total)
  _Float16* wsH  = (_Float16*)d_ws;
  _Float16* Wa1f = wsH;                       // 16*24*512 = 196608
  _Float16* Wa2f = wsH + 196608;              //  8* 8*512 =  32768
  _Float16* Wv1f = wsH + 229376;              // 16*16*512 = 131072
  _Float16* Wv2f = wsH + 360448;              //  8* 8*512 =  32768

  snn_convert_weights<<<(H_DIM * AD + 255) / 256, 256, 0, stream>>>(Wa1, Wa1f, H_DIM, AD);
  snn_convert_weights<<<(E_DIM * H_DIM + 255) / 256, 256, 0, stream>>>(Wa2, Wa2f, E_DIM, H_DIM);
  snn_convert_weights<<<(H_DIM * VD + 255) / 256, 256, 0, stream>>>(Wv1, Wv1f, H_DIM, VD);
  snn_convert_weights<<<(E_DIM * H_DIM + 255) / 256, 256, 0, stream>>>(Wv2, Wv2f, E_DIM, H_DIM);

  snn_kernel<<<B_TOT / ROWS, 256, 0, stream>>>(a_spk, v_spk, Wa1f, ba1, Wa2f, ba2,
                                               Wv1f, bv1, Wv2f, bv2, (float*)d_out);
}
